// CapsuleNet_52553219833863
// MI455X (gfx1250) — matrix operand order repaired and checked by reading, not compiled, not run on hardware
//
#include <hip/hip_runtime.h>
#include <cmath>

typedef __attribute__((ext_vector_type(16))) _Float16 v16h;
typedef __attribute__((ext_vector_type(8)))  float    v8f;

#if __has_builtin(__builtin_amdgcn_wmma_f32_16x16x4_f32)
#define HAVE_WMMA_F32X4 1
typedef __attribute__((ext_vector_type(2))) float v2f;
#endif

#define LN_2PI 1.8378770664093453f

__device__ __forceinline__ float wave_sum(float v) {
#pragma unroll
  for (int off = 16; off > 0; off >>= 1) v += __shfl_xor(v, off, 32);
  return v;
}
__device__ __forceinline__ float wave_max(float v) {
#pragma unroll
  for (int off = 16; off > 0; off >>= 1) v = fmaxf(v, __shfl_xor(v, off, 32));
  return v;
}
__device__ __forceinline__ float fast_rcp(float x) {
  return __builtin_amdgcn_rcpf(x);            // single v_rcp_f32
}
__device__ __forceinline__ float sigmoidf(float s) {
  return fast_rcp(1.f + __expf(-s));          // v_exp + v_rcp
}

// ---------------------------------------------------------------------------
// Stage 1: stem conv 5x5 stride 2 + ReLU -> f16 GEMM A panel (3136 x 32)
// A[m*32 + oc], m = b*196 + oy*14 + ox
// ---------------------------------------------------------------------------
__global__ void stem_conv_f16(const float* __restrict__ x,
                              const float* __restrict__ cw,
                              const float* __restrict__ cb,
                              _Float16* __restrict__ A) {
  int tid = blockIdx.x * blockDim.x + threadIdx.x;
  if (tid >= 16 * 196 * 32) return;
  int oc = tid & 31;
  int m  = tid >> 5;
  int b  = m / 196, pos = m % 196;
  int oy = pos / 14, ox = pos % 14;
  float acc = cb[oc];
#pragma unroll
  for (int ic = 0; ic < 2; ++ic) {
    const float* xp = x + ((b * 2 + ic) * 32 + oy * 2) * 32 + ox * 2;
    const float* wp = cw + (oc * 2 + ic) * 25;
#pragma unroll
    for (int ky = 0; ky < 5; ++ky)
#pragma unroll
      for (int kx = 0; kx < 5; ++kx)
        acc = fmaf(xp[ky * 32 + kx], wp[ky * 5 + kx], acc);
  }
  acc = fmaxf(acc, 0.f);
  A[m * 32 + oc] = (_Float16)acc;
}

// Pack pose_w (512,32) and act_w (32,32) into one f16 B panel: B[k*544 + n]
__global__ void pack_B_f16(const float* __restrict__ pw,
                           const float* __restrict__ aw,
                           _Float16* __restrict__ B) {
  int tid = blockIdx.x * blockDim.x + threadIdx.x;
  if (tid >= 544 * 32) return;
  int n = tid / 32, k = tid % 32;
  float v = (n < 512) ? pw[n * 32 + k] : aw[(n - 512) * 32 + k];
  B[k * 544 + n] = (_Float16)v;
}

// ---------------------------------------------------------------------------
// Stage 2: primary capsules GEMM via v_wmma_f32_16x16x32_f16.
// B weight panel (32x544 f16, 34 KB) staged once per workgroup into LDS with
// 128-bit cooperative copies; fragments built from LDS (ds_load_u16).
// One wave computes a 16x32 slab (2 N-tiles, A fragment reused).
// Fragment layouts per CDNA5 ISA 7.12.2 (wave32).
// ---------------------------------------------------------------------------
__global__ void primary_caps_wmma(const _Float16* __restrict__ A,
                                  const _Float16* __restrict__ B,
                                  const float* __restrict__ pose_b,
                                  const float* __restrict__ act_b,
                                  float* __restrict__ Mprim,   // (16,512,196)
                                  float* __restrict__ aprim) { // (16,32,196)
  __shared__ _Float16 Bl[32 * 544];            // 34816 B of the 320 KB WGP LDS
  for (int t = threadIdx.x; t < (32 * 544 * 2) / 16; t += blockDim.x)
    ((uint4*)Bl)[t] = ((const uint4*)B)[t];
  __syncthreads();

  const int wid = threadIdx.x >> 5;
  const int tp  = blockIdx.x * 4 + wid;
  if (tp >= 196 * 17) return;                  // grid is exact; never taken
  const int tM  = tp / 17;
  const int tN0 = (tp % 17) * 2;
  const int L   = threadIdx.x & 31;
  const int ml  = L & 15, hh = L >> 4, nl = L & 15;

  // A fragment: 16x32 f16. VGPR v: K = (v<4 ? 2v : 16+2(v-4)) + 8*half, row = lane%16.
  const _Float16* Arow = A + (tM * 16 + ml) * 32;
  v16h af;
#pragma unroll
  for (int v = 0; v < 8; ++v) {
    int kb = (v < 4 ? 2 * v : 16 + 2 * (v - 4)) + 8 * hh;
    af[2 * v]     = Arow[kb];
    af[2 * v + 1] = Arow[kb + 1];
  }

#pragma unroll
  for (int bt = 0; bt < 2; ++bt) {
    int tN = tN0 + bt;
    // B fragment from LDS: 32x16 f16. VGPR v: K = (v<4 ? 2v : 16+2(v-4)) + 8*half, col = lane%16.
    v16h bf;
#pragma unroll
    for (int v = 0; v < 8; ++v) {
      int k = (v < 4 ? 2 * v : 16 + 2 * (v - 4)) + 8 * hh;
      bf[2 * v]     = Bl[k * 544 + tN * 16 + nl];
      bf[2 * v + 1] = Bl[(k + 1) * 544 + tN * 16 + nl];
    }
    v8f c = {};
    c = __builtin_amdgcn_wmma_f32_16x16x32_f16(false, af, false, bf,
                                               (short)0, c, false, false);
    // D layout: lane = N col (lanes 16-31 are M+8), VGPR v holds row v (+8).
    int n = tN * 16 + nl;
    if (tN < 32) {                             // wave-uniform pose/act split
#pragma unroll
      for (int v = 0; v < 8; ++v) {
        int m = tM * 16 + v + 8 * hh;
        int bidx = m / 196, pos = m % 196;
        Mprim[(bidx * 512 + n) * 196 + pos] = c[v] + pose_b[n];
      }
    } else {
      int nc = n - 512;
#pragma unroll
      for (int v = 0; v < 8; ++v) {
        int m = tM * 16 + v + 8 * hh;
        int bidx = m / 196, pos = m % 196;
        aprim[(bidx * 32 + nc) * 196 + pos] = sigmoidf(c[v] + act_b[nc]);
      }
    }
  }
}

// ---------------------------------------------------------------------------
// Capsule layers: patch activations ap[b, i=(Bi,kk), p]
// ---------------------------------------------------------------------------
__global__ void compute_ap(const float* __restrict__ a_in, float* __restrict__ ap,
                           int b, int B, int KK, int Kk, int stride,
                           int h, int w, int oh, int ow, int ni, int ohw) {
  long long t = blockIdx.x * (long long)blockDim.x + threadIdx.x;
  long long total = (long long)b * ni * ohw;
  if (t >= total) return;
  int p = (int)(t % ohw);
  int i = (int)((t / ohw) % ni);
  int bb = (int)(t / ((long long)ohw * ni));
  int Bi = i / KK, kk = i % KK;
  int ki = kk / Kk, kj = kk % Kk;
  int oy = p / ow, ox = p % ow;
  int iy = oy * stride + ki, ix = ox * stride + kj;
  ap[t] = a_in[(bb * B + Bi) * h * w + iy * w + ix];
}

// ---------------------------------------------------------------------------
// Pose votes, VALU path (layer 4 / fallback): V = W(4x4) @ Mp(4x4) (+ coords)
// V stored f16 (votes are O(1); halves the dominant HBM stream).
// ---------------------------------------------------------------------------
__global__ void compute_V(const float* __restrict__ M_in, const float* __restrict__ W,
                          _Float16* __restrict__ V,
                          int b, int B, int KK, int Kk, int C, int stride,
                          int h, int w, int oh, int ow, int ni, int nj, int ohw,
                          int share, int add_coord) {
  long long t = blockIdx.x * (long long)blockDim.x + threadIdx.x;
  long long total = (long long)b * ni * nj;
  if (t >= total) return;
  int j = (int)(t % nj);
  int i = (int)((t / nj) % ni);
  int bb = (int)(t / ((long long)nj * ni));
  int c = j / ohw, p = j % ohw;
  int Bi = i / KK, kk = i % KK;
  int ki = kk / Kk, kj = kk % Kk;
  int oy = p / ow, ox = p % ow;
  int iy = oy * stride + ki, ix = ox * stride + kj;
  const float* Wp = share ? (W + (Bi * C + c) * 16)
                          : (W + ((Bi * KK + kk) * C + c) * 16);
  const float* Mb = M_in + ((long long)(bb * B + Bi) * 16) * h * w + iy * w + ix;
  float mp[16];
#pragma unroll
  for (int e = 0; e < 16; ++e) mp[e] = Mb[(long long)e * h * w];
  _Float16* Vp = V + t * 16;
#pragma unroll
  for (int r = 0; r < 4; ++r)
#pragma unroll
    for (int cc = 0; cc < 4; ++cc) {
      float s = 0.f;
#pragma unroll
      for (int k = 0; k < 4; ++k) s = fmaf(Wp[r * 4 + k], mp[k * 4 + cc], s);
      if (add_coord && r == 0) {
        if (cc == 0) s += (float)(stride * ox + ki) / (float)w;
        if (cc == 1) s += (float)(stride * oy + kj) / (float)h;
      }
      Vp[r * 4 + cc] = (_Float16)s;
    }
}

#ifdef HAVE_WMMA_F32X4
// ---------------------------------------------------------------------------
// Pose votes via V_WMMA_F32_16X16X4_F32 (layers 2/3).
// A(16x4)  rows  = (c4, r):  A[m][k]          = W[cbase+(m>>2)][(m&3)*4 + k]
// B(4x16)  cols  = (p4, cc): B[k][n]          = Mp[pbase+(n>>2)][k*4 + (n&3)]
// D(16x16)       : D[(c4*4+r)][(p4*4+cc)]     = V[cbase+c4, pbase+p4][r*4+cc]
// -> all 256 outputs are distinct useful vote elements (zero waste).
// One wave per (b, input capsule i); B fragments for all NPG p-groups are
// pre-gathered into registers (NPG templated so the array stays in VGPRs).
// f32 fragment layout (ISA 7.12.2): VGPR v holds K = v + 2*(lane>=16).
// D stored to V as f16 (b16 stores).
// ---------------------------------------------------------------------------
template <int NPG>
__global__ void compute_V_wmma(const float* __restrict__ M_in,
                               const float* __restrict__ W,
                               _Float16* __restrict__ V,
                               int b, int B, int KK, int Kk, int C, int stride,
                               int h, int w, int oh, int ow, int ni, int nj, int ohw) {
  int bi = blockIdx.x * 8 + (threadIdx.x >> 5);
  if (bi >= b * ni) return;
  int bb = bi / ni, i = bi % ni;
  int Bi = i / KK, kk = i % KK;
  int ki = kk / Kk, kj = kk % Kk;
  const int L  = threadIdx.x & 31;
  const int hh = L >> 4;
  const int nl = L & 15;
  const int m  = L & 15;
  const int hw = h * w;

  // Pre-gather B fragments for all p-groups.
  const float* Mbase = M_in + ((long long)(bb * B + Bi) * 16) * hw;
  v2f bfr[NPG];
#pragma unroll
  for (int pg = 0; pg < NPG; ++pg) {
    int p  = pg * 4 + (nl >> 2);
    int oy = p / ow, ox = p % ow;
    int iy = oy * stride + ki, ix = ox * stride + kj;
    int cc = nl & 3;
#pragma unroll
    for (int v = 0; v < 2; ++v) {
      int e = (v + 2 * hh) * 4 + cc;
      bfr[pg][v] = Mbase[e * hw + iy * w + ix];
    }
  }

  const float* Wbase = W + ((long long)(Bi * KK + kk) * C) * 16;
  for (int cg = 0; cg < (C >> 2); ++cg) {
    v2f af;
#pragma unroll
    for (int v = 0; v < 2; ++v)
      af[v] = Wbase[(cg * 4 + (m >> 2)) * 16 + (m & 3) * 4 + 2 * hh + v];
#pragma unroll
    for (int pg = 0; pg < NPG; ++pg) {
      v8f d = {};
      d = __builtin_amdgcn_wmma_f32_16x16x4_f32(false, af, false, bfr[pg],
                                                (short)0, d, false, false);
      int p   = pg * 4 + (nl >> 2);
      int ccn = nl & 3;
#pragma unroll
      for (int v = 0; v < 8; ++v) {
        int mloc = v + 8 * hh;
        int c = cg * 4 + (mloc >> 2), r = mloc & 3;
        int j = c * ohw + p;
        V[(((long long)bb * ni + i) * nj + j) * 16 + r * 4 + ccn] = (_Float16)d[v];
      }
    }
  }
}
#endif  // HAVE_WMMA_F32X4

__global__ void init_R(float* __restrict__ R, long long n, float v) {
  long long t = blockIdx.x * (long long)blockDim.x + threadIdx.x;
  if (t < n) R[t] = v;
}

// ---------------------------------------------------------------------------
// EM routing M-step: one wave32 per (b, j). coeff / miu / sigma via shfl
// butterfly reductions; emits a_out, miu, 1/sigma, sum(log sigma).
// V rows (32 B f16) loaded as v16h vectors; prefetch next i-row.
// ---------------------------------------------------------------------------
__global__ void em_stats(const float* __restrict__ R, const float* __restrict__ ap,
                         const _Float16* __restrict__ V,
                         const float* __restrict__ beta_a, const float* __restrict__ beta_u,
                         float* __restrict__ miu, float* __restrict__ inv_sigma,
                         float* __restrict__ sls, float* __restrict__ aout,
                         int b, int ni, int nj, int C, int ohw, float lam) {
  int bj = blockIdx.x * 8 + (threadIdx.x >> 5);
  if (bj >= b * nj) return;
  int bb = bj / nj, j = bj % nj;
  int p = j % ohw;
  int lane = threadIdx.x & 31;

  float rl[16];  // R*a cache; ni <= 512 -> <=16 per lane
  int cnt = 0;
  float cf = 0.f;
  for (int i = lane; i < ni; i += 32) {
    float ra = R[((long long)bb * ni + i) * nj + j] *
               ap[((long long)bb * ni + i) * ohw + p];
    rl[cnt++] = ra;
    cf += ra;
  }
  cf = fmaxf(wave_sum(cf), 1e-8f);
  float inv_cf = fast_rcp(cf);

  float mu[16];
#pragma unroll
  for (int e = 0; e < 16; ++e) mu[e] = 0.f;
  cnt = 0;
  for (int i = lane; i < ni; i += 32) {
    float r4 = rl[cnt++] * inv_cf;
    const _Float16* Vp = V + (((long long)bb * ni + i) * nj + j) * 16;
    __builtin_prefetch(Vp + (long long)32 * nj * 16, 0, 1);
    v16h vv = *(const v16h*)Vp;
#pragma unroll
    for (int e = 0; e < 16; ++e) mu[e] = fmaf(r4, (float)vv[e], mu[e]);
  }
#pragma unroll
  for (int e = 0; e < 16; ++e) mu[e] = wave_sum(mu[e]);

  float sg[16];
#pragma unroll
  for (int e = 0; e < 16; ++e) sg[e] = 0.f;
  cnt = 0;
  for (int i = lane; i < ni; i += 32) {
    float r4 = rl[cnt++] * inv_cf;
    const _Float16* Vp = V + (((long long)bb * ni + i) * nj + j) * 16;
    __builtin_prefetch(Vp + (long long)32 * nj * 16, 0, 1);
    v16h vv = *(const v16h*)Vp;
#pragma unroll
    for (int e = 0; e < 16; ++e) {
      float d = (float)vv[e] - mu[e];
      sg[e] = fmaf(r4, d * d, sg[e]);
    }
  }
  float slsum = 0.f;
#pragma unroll
  for (int e = 0; e < 16; ++e) {
    sg[e] = fmaxf(wave_sum(sg[e]), 1e-8f);
    slsum += __logf(sg[e]);
  }
  if (lane == 0) {
    float* mo = miu + ((long long)bb * nj + j) * 16;
    float* io = inv_sigma + ((long long)bb * nj + j) * 16;
#pragma unroll
    for (int e = 0; e < 16; ++e) { mo[e] = mu[e]; io[e] = fast_rcp(sg[e]); }
    sls[bb * nj + j] = slsum;
    float bu = beta_u[j % C], ba = beta_a[j % C];      // faithful j%C indexing
    float cost = cf * (16.f * bu + 0.5f * slsum);
    aout[bb * nj + j] = sigmoidf(lam * (ba - cost));
  }
}

// E-step: one wave32 per (b, i); logits -> in-place softmax over j.
__global__ void em_update_R(float* __restrict__ R, const _Float16* __restrict__ V,
                            const float* __restrict__ miu,
                            const float* __restrict__ inv_sigma,
                            const float* __restrict__ sls,
                            const float* __restrict__ aout,
                            int b, int ni, int nj) {
  int bi = blockIdx.x * 8 + (threadIdx.x >> 5);
  if (bi >= b * ni) return;
  int bb = bi / ni, i = bi % ni;
  int lane = threadIdx.x & 31;
  float* Rrow = R + ((long long)bb * ni + i) * nj;
  const _Float16* Vrow = V + ((long long)bb * ni + i) * nj * 16;

  float mx = -3.4e38f;
  for (int j = lane; j < nj; j += 32) {
    const _Float16* Vp = Vrow + (long long)j * 16;
    __builtin_prefetch(Vp + 32 * 16, 0, 1);
    v16h vv = *(const v16h*)Vp;
    const float* mp = miu + ((long long)bb * nj + j) * 16;
    const float* ip = inv_sigma + ((long long)bb * nj + j) * 16;
    float dot = 0.f;
#pragma unroll
    for (int e = 0; e < 16; ++e) {
      float d = (float)vv[e] - mp[e];
      dot = fmaf(d * d, ip[e], dot);
    }
    float logp = -0.5f * (16.f * LN_2PI + sls[bb * nj + j] + dot);
    float xv = aout[bb * nj + j] * logp;
    Rrow[j] = xv;
    mx = fmaxf(mx, xv);
  }
  mx = wave_max(mx);
  float sum = 0.f;
  for (int j = lane; j < nj; j += 32) {
    float e = __expf(Rrow[j] - mx);
    Rrow[j] = e;
    sum += e;
  }
  sum = wave_sum(sum);
  float inv = fast_rcp(sum);
  for (int j = lane; j < nj; j += 32) Rrow[j] *= inv;
}

// Final relayout per layer: a -> (b*C, ohw); M -> faithful (b,16,C,ohw) flat.
__global__ void em_output(const float* __restrict__ aout, const float* __restrict__ miu,
                          float* __restrict__ a_next, float* __restrict__ M_next,
                          int b, int nj, int C, int ohw, int write_M) {
  int t = blockIdx.x * blockDim.x + threadIdx.x;
  if (t >= b * nj) return;
  int bb = t / nj, j = t % nj;
  int c = j / ohw, p = j % ohw;
  a_next[(bb * C + c) * ohw + p] = aout[t];
  if (write_M) {
    const float* mp = miu + (long long)t * 16;
#pragma unroll
    for (int e = 0; e < 16; ++e)
      M_next[(((long long)bb * 16 + e) * C + c) * ohw + p] = mp[e];
  }
}

// ---------------------------------------------------------------------------
extern "C" void kernel_launch(void* const* d_in, const int* in_sizes, int n_in,
                              void* d_out, int out_size, void* d_ws, size_t ws_size,
                              hipStream_t stream) {
  (void)in_sizes; (void)n_in; (void)out_size; (void)ws_size;
  const float* x      = (const float*)d_in[0];
  const float* conv_w = (const float*)d_in[1];
  const float* conv_b = (const float*)d_in[2];
  const float* pose_w = (const float*)d_in[3];
  const float* pose_b = (const float*)d_in[4];
  const float* act_w  = (const float*)d_in[5];
  const float* act_b  = (const float*)d_in[6];
  const float* W2  = (const float*)d_in[7];
  const float* ba2 = (const float*)d_in[8];
  const float* bu2 = (const float*)d_in[9];
  const float* W3  = (const float*)d_in[10];
  const float* ba3 = (const float*)d_in[11];
  const float* bu3 = (const float*)d_in[12];
  const float* W4  = (const float*)d_in[13];
  const float* ba4 = (const float*)d_in[14];
  const float* bu4 = (const float*)d_in[15];

  const int b = 16;
  char* base = (char*)d_ws;
  size_t off = 0;
  auto alloc = [&](size_t bytes) -> void* {
    off = (off + 255) & ~(size_t)255;
    void* p = base + off;
    off += bytes;
    return p;
  };
  _Float16* A_h = (_Float16*)alloc((size_t)3136 * 32 * 2);
  _Float16* B_h = (_Float16*)alloc((size_t)544 * 32 * 2);
  float* Mprim = (float*)alloc((size_t)b * 512 * 196 * 4);
  float* aprim = (float*)alloc((size_t)b * 32 * 196 * 4);
  float* a2 = (float*)alloc((size_t)b * 32 * 36 * 4);
  float* M2 = (float*)alloc((size_t)b * 16 * 32 * 36 * 4);
  float* a3 = (float*)alloc((size_t)b * 32 * 16 * 4);
  float* M3 = (float*)alloc((size_t)b * 16 * 32 * 16 * 4);
  _Float16* Vbuf = (_Float16*)alloc((size_t)b * 288 * 1152 * 16 * 2);  // 170 MB peak
  float* Rbuf  = (float*)alloc((size_t)b * 288 * 1152 * 4);
  float* apbuf = (float*)alloc((size_t)b * 512 * 36 * 4);
  float* miu = (float*)alloc((size_t)b * 1152 * 16 * 4);
  float* isg = (float*)alloc((size_t)b * 1152 * 16 * 4);
  float* sls = (float*)alloc((size_t)b * 1152 * 4);
  float* aou = (float*)alloc((size_t)b * 1152 * 4);

  // Stage 1-2: stem conv + primary caps WMMA GEMM
  stem_conv_f16<<<(b * 196 * 32 + 255) / 256, 256, 0, stream>>>(x, conv_w, conv_b, A_h);
  pack_B_f16<<<(544 * 32 + 255) / 256, 256, 0, stream>>>(pose_w, act_w, B_h);
  primary_caps_wmma<<<(196 * 17 + 3) / 4, 128, 0, stream>>>(A_h, B_h, pose_b, act_b,
                                                            Mprim, aprim);

  struct LCfg {
    const float *a_in, *M_in, *W, *ba, *bu;
    int B, KK, Kk, C, stride, h, w, oh, ow, share, addc;
    float *a_o, *M_o;
    int write_M;
  };
  LCfg Ls[3] = {
    { aprim, Mprim, W2, ba2, bu2, 32,  9, 3, 32, 2, 14, 14, 6, 6, 0, 0, a2, M2, 1 },
    { a2,    M2,    W3, ba3, bu3, 32,  9, 3, 32, 1,  6,  6, 4, 4, 0, 0, a3, M3, 1 },
    { a3,    M3,    W4, ba4, bu4, 32, 16, 4,  5, 1,  4,  4, 1, 1, 1, 1,
      (float*)d_out, nullptr, 0 },
  };

  for (int l = 0; l < 3; ++l) {
    const LCfg& L = Ls[l];
    const int ni = L.B * L.KK, ohw = L.oh * L.ow, nj = L.C * ohw;

    {
      long long total = (long long)b * ni * ohw;
      compute_ap<<<(unsigned)((total + 255) / 256), 256, 0, stream>>>(
          L.a_in, apbuf, b, L.B, L.KK, L.Kk, L.stride, L.h, L.w, L.oh, L.ow, ni, ohw);
    }

    bool wmma_v = false;
#ifdef HAVE_WMMA_F32X4
    if (l == 0) {
      compute_V_wmma<9><<<(b * ni + 7) / 8, 256, 0, stream>>>(
          L.M_in, L.W, Vbuf, b, L.B, L.KK, L.Kk, L.C, L.stride,
          L.h, L.w, L.oh, L.ow, ni, nj, ohw);
      wmma_v = true;
    } else if (l == 1) {
      compute_V_wmma<4><<<(b * ni + 7) / 8, 256, 0, stream>>>(
          L.M_in, L.W, Vbuf, b, L.B, L.KK, L.Kk, L.C, L.stride,
          L.h, L.w, L.oh, L.ow, ni, nj, ohw);
      wmma_v = true;
    }
#endif
    if (!wmma_v) {
      long long total = (long long)b * ni * nj;
      compute_V<<<(unsigned)((total + 255) / 256), 256, 0, stream>>>(
          L.M_in, L.W, Vbuf, b, L.B, L.KK, L.Kk, L.C, L.stride,
          L.h, L.w, L.oh, L.ow, ni, nj, ohw, L.share, L.addc);
    }
    {
      long long total = (long long)b * ni * nj;
      init_R<<<(unsigned)((total + 255) / 256), 256, 0, stream>>>(
          Rbuf, total, 1.f / (float)nj);
    }
    for (int t = 0; t < 3; ++t) {
      float lam = 0.01f * (1.f - powf(0.95f, (float)(t + 1)));
      int nbj = b * nj;
      em_stats<<<(nbj + 7) / 8, 256, 0, stream>>>(
          Rbuf, apbuf, Vbuf, L.ba, L.bu, miu, isg, sls, aou,
          b, ni, nj, L.C, ohw, lam);
      if (t < 2) {
        int nbi = b * ni;
        em_update_R<<<(nbi + 7) / 8, 256, 0, stream>>>(
            Rbuf, Vbuf, miu, isg, sls, aou, b, ni, nj);
      }
    }
    em_output<<<(b * nj + 255) / 256, 256, 0, stream>>>(
        aou, miu, L.a_o, L.M_o, b, nj, L.C, ohw, L.write_M);
  }
}
